// SelfAttentiveLBLBiLMV2_21345987461681
// MI455X (gfx1250) — matrix operand order repaired and checked by reading, not compiled, not run on hardware
//
#include <hip/hip_runtime.h>

// ---------------------------------------------------------------------------
// Problem constants (from the reference)
// ---------------------------------------------------------------------------
namespace {

constexpr int kWidth   = 16;
constexpr int kHid     = 512;
constexpr int kHeads   = 8;
constexpr int kDk      = 64;                  // kHid / kHeads
constexpr int kLayers  = 2;
constexpr int kHighway = 2;
constexpr int kB       = 4;
constexpr int kS       = 1024;
constexpr int kPad     = kS + 2 * kWidth;     // 1056
constexpr int kBPad    = kB * kPad;           // 4224  (multiple of 128)
constexpr int kBS      = kB * kS;             // 4096  (multiple of 128)
constexpr int kTaps    = kWidth + 1;          // 17
constexpr int kWin     = kWidth + 2;          // 18-wide attention band

typedef __attribute__((ext_vector_type(8)))  __bf16 bf16x8;
typedef __attribute__((ext_vector_type(16))) __bf16 bf16x16;
typedef __attribute__((ext_vector_type(8)))  float  f32x8;

// ---------------------------------------------------------------------------
// f32 -> bf16 convert (weights, done once per launch)
// ---------------------------------------------------------------------------
__global__ __launch_bounds__(256) void convert_bf16_kernel(
    const float* __restrict__ src, __bf16* __restrict__ dst, int n) {
  int i = blockIdx.x * blockDim.x + threadIdx.x;
  if (i < n) dst[i] = (__bf16)src[i];
}

// ---------------------------------------------------------------------------
// Build padded sequence [B, PAD, HID] in bf16:
//   rows [0,16)      = fw_pad[l]   (broadcast over batch)
//   rows [16,16+S)   = state (f32, [B,S,HID])
//   rows [16+S,PAD)  = bw_pad[l]
// ---------------------------------------------------------------------------
__global__ __launch_bounds__(256) void build_padded_kernel(
    const float* __restrict__ state, const float* __restrict__ fpad,
    const float* __restrict__ bpad, __bf16* __restrict__ xp) {
  int idx = blockIdx.x * blockDim.x + threadIdx.x;
  if (idx >= kB * kPad * kHid) return;
  int h  = idx % kHid;
  int bp = idx / kHid;
  int p  = bp % kPad;
  int b  = bp / kPad;
  float v;
  if (p < kWidth)              v = fpad[p * kHid + h];
  else if (p >= kWidth + kS)   v = bpad[(p - kWidth - kS) * kHid + h];
  else                         v = state[((size_t)b * kS + (p - kWidth)) * kHid + h];
  xp[idx] = (__bf16)v;
}

// ---------------------------------------------------------------------------
// WMMA GEMM:  Y[M,N] = X[M,K=512] * W[N,K=512]^T + bias[N]   (bf16 in, f32 out)
//
// Block = 256 threads = 8 waves; block tile 128(M) x 64(N).
// The 64x512 bf16 weight strip shared by all 8 waves is staged into LDS once
// per block with CDNA5 async global->LDS copies (ASYNCcnt). The K-loop is
// unrolled x2 with ping-pong A-fragment registers (no cross-buffer v_movs ->
// no WMMA->VALU hazard NOPs), and each half-step batches its 4 B-fragment
// ds_loads ahead of its 4 v_wmma_f32_16x16x32_bf16 issues.
// ---------------------------------------------------------------------------
__global__ __launch_bounds__(256) void gemm_bias_kernel(
    const __bf16* __restrict__ X, const __bf16* __restrict__ W,
    const float* __restrict__ bias, float* __restrict__ Y, int M, int N) {
  constexpr int K = kHid;  // 512 for every GEMM in this model
  __shared__ __bf16 sW[64 * K];  // 64 KB weight strip for this block's N range

  const int tid  = threadIdx.x;
  const int lane = tid & 31;
  const int wave = tid >> 5;
  const int m0   = blockIdx.y * 128 + wave * 16;
  const int n0   = blockIdx.x * 64;

  // ---- async-stage W[n0:n0+64, 0:512] into LDS (64 KB, 16B per thread/iter)
  {
    const __bf16* wsrc = W + (size_t)n0 * K;
    unsigned lds_base = (unsigned)(uintptr_t)(&sW[0]);  // generic->LDS: addr[31:0]
#pragma unroll
    for (int it = 0; it < 16; ++it) {
      const int elem = (it * 256 + tid) * 8;          // 8 bf16 = 16 bytes
      unsigned laddr = lds_base + (unsigned)(elem * 2);
      unsigned long long gaddr = (unsigned long long)(uintptr_t)(wsrc + elem);
      asm volatile("global_load_async_to_lds_b128 %0, %1, off"
                   :: "v"(laddr), "v"(gaddr)
                   : "memory");
    }
    asm volatile("s_wait_asynccnt 0x0" ::: "memory");
    __syncthreads();
  }

  // A fragment addressing (this code's contiguous k order; B uses the same):
  //   lanes 0-15 : row = lane,    K = {0..15}
  //   lanes16-31 : row = lane-16, K = {16..31}
  const int arow  = lane & 15;
  const int akoff = (lane >> 4) << 4;   // 0 or 16
  // B fragment addressing (16-bit B 32x16 layout):
  //   lanes 0-15 : col = lane,    K = {0..15};  lanes16-31: K = {16..31}
  const int bcol  = lane & 15;
  const int bkoff = (lane >> 4) << 4;   // 0 or 16

  const __bf16* __restrict__ xa = X + (size_t)(m0 + arow) * K + akoff;
  const __bf16* __restrict__ sb = &sW[bcol * K + bkoff];

  f32x8 acc[4] = {};

  // ping-pong A fragments: buffer 0 covers kk, buffer 1 covers kk+32
  bf16x8 alo0 = *(const bf16x8*)(xa);
  bf16x8 ahi0 = *(const bf16x8*)(xa + 8);

#pragma unroll 1
  for (int kk = 0; kk < K; kk += 64) {
    // ---- half-step 0: consume buffer 0 at kk, prefetch buffer 1 (kk+32)
    bf16x8 alo1 = *(const bf16x8*)(xa + kk + 32);
    bf16x8 ahi1 = *(const bf16x8*)(xa + kk + 40);

    bf16x16 b0[4];
#pragma unroll
    for (int nb = 0; nb < 4; ++nb)
      b0[nb] = *(const bf16x16*)(sb + nb * 16 * K + kk);

    bf16x16 a0;
#pragma unroll
    for (int t = 0; t < 8; ++t) { a0[t] = alo0[t]; a0[8 + t] = ahi0[t]; }
#pragma unroll
    for (int nb = 0; nb < 4; ++nb)
      acc[nb] = __builtin_amdgcn_wmma_f32_16x16x32_bf16(
          false, a0, false, b0[nb], (short)0, acc[nb], false, false);

    // ---- half-step 1: consume buffer 1 at kk+32, prefetch buffer 0 (kk+64)
    if (kk + 64 < K) {
      alo0 = *(const bf16x8*)(xa + kk + 64);
      ahi0 = *(const bf16x8*)(xa + kk + 72);
    }

    bf16x16 b1[4];
#pragma unroll
    for (int nb = 0; nb < 4; ++nb)
      b1[nb] = *(const bf16x16*)(sb + nb * 16 * K + kk + 32);

    bf16x16 a1;
#pragma unroll
    for (int t = 0; t < 8; ++t) { a1[t] = alo1[t]; a1[8 + t] = ahi1[t]; }
#pragma unroll
    for (int nb = 0; nb < 4; ++nb)
      acc[nb] = __builtin_amdgcn_wmma_f32_16x16x32_bf16(
          false, a1, false, b1[nb], (short)0, acc[nb], false, false);
  }

  // C/D layout: VGPR r -> M = m0 + r + 8*(lane>=16); N = n0 + nb*16 + (lane&15)
  const int mbase = m0 + ((lane >> 4) << 3);
  const int cbase = n0 + (lane & 15);
#pragma unroll
  for (int nb = 0; nb < 4; ++nb) {
    const int c  = cbase + nb * 16;
    const float bv = bias[c];
#pragma unroll
    for (int r = 0; r < 8; ++r) {
      Y[(size_t)(mbase + r) * N + c] = acc[nb][r] + bv;
    }
  }
}

// ---------------------------------------------------------------------------
// Banded softmax attention. One wave32 per (b, h, i); lane owns 2 of DK=64
// dims. Window: fwd j in [i-17, i], bwd j in [i, i+17] (clamped to [0,PAD)).
// Emits bf16 context directly for the out-projection GEMM.
// ---------------------------------------------------------------------------
__global__ __launch_bounds__(256) void band_attn_kernel(
    const float* __restrict__ Q, const float* __restrict__ Kt,
    const float* __restrict__ V, __bf16* __restrict__ O, int forward) {
  const int gwave = (int)((blockIdx.x * blockDim.x + threadIdx.x) >> 5);
  const int lane  = threadIdx.x & 31;
  if (gwave >= kB * kHeads * kPad) return;
  const int i  = gwave % kPad;
  const int bh = gwave / kPad;
  const int h  = bh % kHeads;
  const int b  = bh / kHeads;

  const int d0 = lane * 2;
  const size_t base = (size_t)b * kPad * kHid;
  const float* q = Q + base + (size_t)i * kHid + h * kDk;
  const float q0 = q[d0], q1 = q[d0 + 1];

  int jlo, jhi;
  if (forward) { jlo = i - kWidth - 1; if (jlo < 0) jlo = 0; jhi = i; }
  else         { jlo = i; jhi = i + kWidth + 1; if (jhi > kPad - 1) jhi = kPad - 1; }
  const int cnt = jhi - jlo + 1;  // wave-uniform, <= 18

  float s[kWin];
  for (int t = 0; t < cnt; ++t) {
    const float* kp = Kt + base + (size_t)(jlo + t) * kHid + h * kDk;
    float p = q0 * kp[d0] + q1 * kp[d0 + 1];
    p += __shfl_xor(p, 16, 32);
    p += __shfl_xor(p,  8, 32);
    p += __shfl_xor(p,  4, 32);
    p += __shfl_xor(p,  2, 32);
    p += __shfl_xor(p,  1, 32);
    s[t] = p * 0.125f;  // 1/sqrt(DK=64)
  }
  float m = -3.0e38f;
  for (int t = 0; t < cnt; ++t) m = fmaxf(m, s[t]);
  float sum = 0.f;
  for (int t = 0; t < cnt; ++t) { s[t] = __expf(s[t] - m); sum += s[t]; }
  const float inv = 1.0f / sum;

  float o0 = 0.f, o1 = 0.f;
  for (int t = 0; t < cnt; ++t) {
    const float* vp = V + base + (size_t)(jlo + t) * kHid + h * kDk;
    const float w = s[t] * inv;
    o0 += w * vp[d0];
    o1 += w * vp[d0 + 1];
  }
  __bf16* op = O + base + (size_t)i * kHid + h * kDk;
  op[d0]     = (__bf16)o0;
  op[d0 + 1] = (__bf16)o1;
}

// ---------------------------------------------------------------------------
// out[t] = fa[16+t] + sum_{k=0..16} w[k] * fa[offset+k+t]   (per b, hidden ch)
// Writes the new direction state in f32 + bf16 (highway GEMM input).
// ---------------------------------------------------------------------------
__global__ __launch_bounds__(256) void rel_combine_kernel(
    const float* __restrict__ FA, const float* __restrict__ w,
    float* __restrict__ state, __bf16* __restrict__ state_bf, int offset) {
  int idx = blockIdx.x * blockDim.x + threadIdx.x;
  if (idx >= kB * kS * kHid) return;
  int h  = idx % kHid;
  int bt = idx / kHid;
  int t  = bt % kS;
  int b  = bt / kS;
  const float* fab = FA + (size_t)b * kPad * kHid + h;
  float acc = fab[(size_t)(kWidth + t) * kHid];
#pragma unroll
  for (int k = 0; k < kTaps; ++k)
    acc += w[k] * fab[(size_t)(offset + k + t) * kHid];
  state[idx]    = acc;
  state_bf[idx] = (__bf16)acc;
}

// ---------------------------------------------------------------------------
// Highway gating: x = g*x + (1-g)*relu(nl),  g = sigmoid(proj[:,H:])
// ---------------------------------------------------------------------------
__global__ __launch_bounds__(256) void highway_gate_kernel(
    const float* __restrict__ proj, float* __restrict__ state,
    __bf16* __restrict__ state_bf) {
  int idx = blockIdx.x * blockDim.x + threadIdx.x;
  if (idx >= kB * kS * kHid) return;
  int h  = idx % kHid;
  int bt = idx / kHid;
  float x  = state[idx];
  float nl = proj[(size_t)bt * (2 * kHid) + h];
  nl = fmaxf(nl, 0.f);
  float g = proj[(size_t)bt * (2 * kHid) + kHid + h];
  g = 1.0f / (1.0f + __expf(-g));
  float y = g * x + (1.0f - g) * nl;
  state[idx]    = y;
  state_bf[idx] = (__bf16)y;
}

// ---------------------------------------------------------------------------
// out[l, b, t, 0:512] = f ; out[l, b, t, 512:1024] = bk
// ---------------------------------------------------------------------------
__global__ __launch_bounds__(256) void write_out_kernel(
    const float* __restrict__ f, const float* __restrict__ bk,
    float* __restrict__ out) {
  int idx = blockIdx.x * blockDim.x + threadIdx.x;
  if (idx >= kB * kS * kHid) return;
  int h  = idx % kHid;
  int bt = idx / kHid;
  out[(size_t)bt * (2 * kHid) + h]        = f[idx];
  out[(size_t)bt * (2 * kHid) + kHid + h] = bk[idx];
}

}  // namespace

// ---------------------------------------------------------------------------
// Host orchestration (graph-capture safe: stream-ordered launches only)
// ---------------------------------------------------------------------------
extern "C" void kernel_launch(void* const* d_in, const int* in_sizes, int n_in,
                              void* d_out, int out_size, void* d_ws, size_t ws_size,
                              hipStream_t stream) {
  (void)in_sizes; (void)n_in; (void)out_size; (void)ws_size;

  const float* inputs    = (const float*)d_in[0];
  // d_in[1] = masks (unused by the reference body)
  const float* fw_lin_w  = (const float*)d_in[2];
  const float* fw_lin_b  = (const float*)d_in[3];
  const float* bw_lin_w  = (const float*)d_in[4];
  const float* bw_lin_b  = (const float*)d_in[5];
  const float* fw_hw_w   = (const float*)d_in[6];
  const float* fw_hw_b   = (const float*)d_in[7];
  const float* bw_hw_w   = (const float*)d_in[8];
  const float* bw_hw_b   = (const float*)d_in[9];
  const float* fw_pad    = (const float*)d_in[10];
  const float* bw_pad    = (const float*)d_in[11];
  const float* fw_rel    = (const float*)d_in[12];
  const float* bw_rel    = (const float*)d_in[13];
  float* out = (float*)d_out;

  // ---- bump allocator on d_ws ----
  char* wp = (char*)d_ws;
  auto alloc = [&](size_t bytes) -> void* {
    void* r = (void*)wp;
    wp += (bytes + 255) & ~(size_t)255;
    return r;
  };
  const size_t nLin  = (size_t)kLayers * 4 * kHid * kHid;            // 2,097,152
  const size_t nHw   = (size_t)kLayers * kHighway * 2 * kHid * kHid; // 2,097,152
  const size_t nPadA = (size_t)kBPad * kHid;                         // 2,162,688
  const size_t nSt   = (size_t)kBS * kHid;                           // 2,097,152

  __bf16* linf_bf = (__bf16*)alloc(sizeof(__bf16) * nLin);
  __bf16* linb_bf = (__bf16*)alloc(sizeof(__bf16) * nLin);
  __bf16* hwf_bf  = (__bf16*)alloc(sizeof(__bf16) * nHw);
  __bf16* hwb_bf  = (__bf16*)alloc(sizeof(__bf16) * nHw);
  __bf16* xpad_bf = (__bf16*)alloc(sizeof(__bf16) * nPadA);
  float*  qbuf    = (float*)alloc(sizeof(float) * nPadA);
  float*  kbuf    = (float*)alloc(sizeof(float) * nPadA);
  float*  vbuf    = (float*)alloc(sizeof(float) * nPadA);
  __bf16* obuf_bf = (__bf16*)alloc(sizeof(__bf16) * nPadA);
  float*  fabuf   = (float*)alloc(sizeof(float) * nPadA);
  float*  st_f    = (float*)alloc(sizeof(float) * nSt);
  float*  st_b    = (float*)alloc(sizeof(float) * nSt);
  __bf16* st_f_bf = (__bf16*)alloc(sizeof(__bf16) * nSt);
  __bf16* st_b_bf = (__bf16*)alloc(sizeof(__bf16) * nSt);
  // highway projection [B*S, 1024] f32 (16.78 MB) aliases q+k (17.3 MB, dead
  // by the time the highway runs).
  float* proj = qbuf;

  const dim3 blk(256);
  auto gblocks = [](size_t n) { return dim3((unsigned)((n + 255) / 256)); };

  // ---- weight conversion f32 -> bf16 (once per launch, deterministic) ----
  convert_bf16_kernel<<<gblocks(nLin), blk, 0, stream>>>(fw_lin_w, linf_bf, (int)nLin);
  convert_bf16_kernel<<<gblocks(nLin), blk, 0, stream>>>(bw_lin_w, linb_bf, (int)nLin);
  convert_bf16_kernel<<<gblocks(nHw),  blk, 0, stream>>>(fw_hw_w,  hwf_bf,  (int)nHw);
  convert_bf16_kernel<<<gblocks(nHw),  blk, 0, stream>>>(bw_hw_w,  hwb_bf,  (int)nHw);

  auto gemm = [&](const __bf16* X, const __bf16* W, const float* bias,
                  float* Y, int M, int N) {
    dim3 grid(N / 64, M / 128);
    gemm_bias_kernel<<<grid, blk, 0, stream>>>(X, W, bias, Y, M, N);
  };

  for (int l = 0; l < kLayers; ++l) {
    const float* fp = fw_pad + (size_t)l * kWidth * kHid;
    const float* bp = bw_pad + (size_t)l * kWidth * kHid;

    for (int dir = 0; dir < 2; ++dir) {
      const bool fwd = (dir == 0);
      const float*  src   = (l == 0) ? inputs : (fwd ? st_f : st_b);
      const __bf16* linW  = (fwd ? linf_bf : linb_bf) + (size_t)l * 4 * kHid * kHid;
      const float*  linB  = (fwd ? fw_lin_b : bw_lin_b) + (size_t)l * 4 * kHid;
      const __bf16* hwW   = (fwd ? hwf_bf : hwb_bf) + (size_t)l * kHighway * 2 * kHid * kHid;
      const float*  hwB   = (fwd ? fw_hw_b : bw_hw_b) + (size_t)l * kHighway * 2 * kHid;
      const float*  relW  = (fwd ? fw_rel : bw_rel) + (size_t)l * kTaps;
      float*        st    = fwd ? st_f : st_b;
      __bf16*       st_bf = fwd ? st_f_bf : st_b_bf;

      // 1) padded input (bf16)
      build_padded_kernel<<<gblocks(nPadA), blk, 0, stream>>>(src, fp, bp, xpad_bf);

      // 2) Q, K, V projections (WMMA)
      gemm(xpad_bf, linW + 0 * (size_t)kHid * kHid, linB + 0 * kHid, qbuf, kBPad, kHid);
      gemm(xpad_bf, linW + 1 * (size_t)kHid * kHid, linB + 1 * kHid, kbuf, kBPad, kHid);
      gemm(xpad_bf, linW + 2 * (size_t)kHid * kHid, linB + 2 * kHid, vbuf, kBPad, kHid);

      // 3) banded attention -> bf16 context
      {
        const size_t waves = (size_t)kB * kHeads * kPad;  // 33792
        band_attn_kernel<<<dim3((unsigned)(waves / 8)), blk, 0, stream>>>(
            qbuf, kbuf, vbuf, obuf_bf, fwd ? 1 : 0);
      }

      // 4) output projection (WMMA)
      gemm(obuf_bf, linW + 3 * (size_t)kHid * kHid, linB + 3 * kHid, fabuf, kBPad, kHid);

      // 5) residual + 17-tap relative-position combine -> direction state
      rel_combine_kernel<<<gblocks(nSt), blk, 0, stream>>>(
          fabuf, relW, st, st_bf, fwd ? 0 : kWidth);

      // 6) two highway layers (WMMA GEMM N=1024 + gating)
      for (int i = 0; i < kHighway; ++i) {
        gemm(st_bf, hwW + (size_t)i * 2 * kHid * kHid, hwB + (size_t)i * 2 * kHid,
             proj, kBS, 2 * kHid);
        highway_gate_kernel<<<gblocks(nSt), blk, 0, stream>>>(proj, st, st_bf);
      }
    }

    // 7) out[l] = concat(f, bk)
    write_out_kernel<<<gblocks(nSt), blk, 0, stream>>>(
        st_f, st_b, out + (size_t)l * kBS * 2 * kHid);
  }
}
